// MultiHeadAttention_23751169147472
// MI455X (gfx1250) — compile-verified, matrix operand order checked by reading, not run on hardware
//
#include <hip/hip_runtime.h>
#include <cstdint>
#include <cstddef>

// ---------------------------------------------------------------------------
// MHA block for B=2, L=2048, D=1024, H=16, DPH=64 on gfx1250 (MI455X).
// bf16 WMMA (v_wmma_f32_16x16x32_bf16) for all 6 GEMMs; softmax/LN in f32.
// The reference's flat reshape makes every per-head matrix a contiguous
// [2048 x 64] slice, so "heads" are just a batch stride of 131072 elements.
// LDS staging uses async global->LDS copies (ASYNCcnt) with double buffering
// when the toolchain exposes the gfx1250 builtin; falls back to b128 loads.
// ---------------------------------------------------------------------------

typedef __attribute__((ext_vector_type(16))) __bf16 v16bf;
typedef __attribute__((ext_vector_type(8)))  __bf16 v8bf;
typedef __attribute__((ext_vector_type(8)))  float  v8f;
typedef int v4i __attribute__((vector_size(16)));

#if defined(__HIP_DEVICE_COMPILE__) && __has_builtin(__builtin_amdgcn_global_load_async_to_lds_b128)
#define HAVE_ASYNC_LDS 1
#else
#define HAVE_ASYNC_LDS 0
#endif

static __device__ __forceinline__ void wait_async0() {
#if defined(__HIP_DEVICE_COMPILE__)
#if __has_builtin(__builtin_amdgcn_s_wait_asynccnt)
  __builtin_amdgcn_s_wait_asynccnt(0);
#else
  asm volatile("s_wait_asynccnt 0x0" ::: "memory");
#endif
#endif
}

static __device__ __forceinline__ void async_cp16(const void* gsrc, void* ldst) {
#if HAVE_ASYNC_LDS
  typedef __attribute__((address_space(1))) v4i* gp128;  // v4i addrspace(1)*
  typedef __attribute__((address_space(3))) v4i* lp128;  // v4i addrspace(3)*
  __builtin_amdgcn_global_load_async_to_lds_b128((gp128)gsrc, (lp128)ldst, 0, 0);
#else
  *(uint4*)ldst = *(const uint4*)gsrc;
#endif
}

static __device__ __forceinline__ unsigned short f2bf(float f) {
  unsigned int u = __float_as_uint(f);
  u += 0x7FFFu + ((u >> 16) & 1u);   // round-to-nearest-even
  return (unsigned short)(u >> 16);
}

// ---------------- elementwise f32 -> bf16 ----------------------------------
__global__ __launch_bounds__(256) void cvt_f32_to_bf16(
    const float* __restrict__ in, unsigned short* __restrict__ out, int n4) {
  int i = blockIdx.x * blockDim.x + threadIdx.x;
  if (i >= n4) return;
  float4 v = ((const float4*)in)[i];
  ((ushort4*)out)[i] = make_ushort4(f2bf(v.x), f2bf(v.y), f2bf(v.z), f2bf(v.w));
}

// ---------------- per-head transpose of V: [32][2048][64] -> [32][64][2048] -
__global__ __launch_bounds__(256) void transpose_heads_v(
    const unsigned short* __restrict__ V, unsigned short* __restrict__ Vt) {
  int idx = blockIdx.x * blockDim.x + threadIdx.x;  // < 4194304
  int bh  = idx >> 17;
  int rem = idx & 131071;
  int j = rem >> 6, d = rem & 63;
  Vt[(size_t)bh * 131072 + (size_t)d * 2048 + j] = V[idx];
}

// ---------------- generic WMMA GEMM: Y = A * B^T (+bias)(+resid) -----------
// A: [M x K] (bf16, or f32 converted on the fly when A_F32)
// B: [N x K] row-major bf16 (== B-operand lane layout: lane=n, regs=k)
// OUT_MODE: 0 = bf16 (+bias if non-null), 1 = f32 * scale, 2 = f32+bias+resid
template<bool A_F32, int BN, int OUT_MODE>
__global__ __launch_bounds__(256) void gemm_wmma(
    const void* __restrict__ Aglob,
    const unsigned short* __restrict__ Bglob,
    const float* __restrict__ bias,
    const float* __restrict__ resid,
    void* __restrict__ Cglob,
    int K, int lda, int ldb, int ldc,
    long long strideA, long long strideB, long long strideC,
    float scale)
{
  constexpr int BM = 128;
  constexpr int BK = 32;
  constexpr int WAVES_N = (BN == 128) ? 4 : 2;
  constexpr int WAVES_M = 8 / WAVES_N;
  constexpr int TM = BM / (WAVES_M * 16);
  constexpr int TN = BN / (WAVES_N * 16);

  __shared__ __align__(16) unsigned short sA[2][BM * BK];
  __shared__ __align__(16) unsigned short sB[2][BN * BK];

  const int tid  = threadIdx.x;
  const int lane = tid & 31;
  const int wave = tid >> 5;
  const int wm   = wave / WAVES_N;
  const int wn   = wave % WAVES_N;
  const int z      = blockIdx.z;
  const int mBlock = blockIdx.y * BM;
  const int nBlock = blockIdx.x * BN;

  const unsigned short* Ab = A_F32 ? nullptr
      : (const unsigned short*)Aglob + (size_t)z * strideA;
  const float* Af = A_F32 ? (const float*)Aglob + (size_t)z * strideA : nullptr;
  const unsigned short* Bb = Bglob + (size_t)z * strideB;

  v8f acc[TM][TN];
  #pragma unroll
  for (int i = 0; i < TM; ++i)
    #pragma unroll
    for (int j = 0; j < TN; ++j) { v8f zz = {}; acc[i][j] = zz; }

  const int lrow  = lane & 15;
  const int lhalf = lane >> 4;
  const int kb    = lhalf * 8;      // K sub-offset within 32-wide slab

  // stage one [rows x 32] bf16 slab into LDS buffer `buf` at K offset k0
  auto stageA = [&](int buf, int k0) {
    if constexpr (A_F32) {
      #pragma unroll
      for (int it = 0; it < (BM * 4) / 256; ++it) {
        int i = tid + it * 256;
        int row = i >> 2, ch = (i & 3) * 8;
        const float* src = Af + (size_t)(mBlock + row) * lda + k0 + ch;
        float4 x0 = ((const float4*)src)[0];
        float4 x1 = ((const float4*)src)[1];
        *(ushort4*)&sA[buf][row * BK + ch] =
            make_ushort4(f2bf(x0.x), f2bf(x0.y), f2bf(x0.z), f2bf(x0.w));
        *(ushort4*)&sA[buf][row * BK + ch + 4] =
            make_ushort4(f2bf(x1.x), f2bf(x1.y), f2bf(x1.z), f2bf(x1.w));
        if (k0 + BK < K) __builtin_prefetch(src + BK, 0, 1);
      }
    } else {
      #pragma unroll
      for (int it = 0; it < (BM * 4) / 256; ++it) {
        int i = tid + it * 256;
        int row = i >> 2, ch = (i & 3) * 8;
        async_cp16(Ab + (size_t)(mBlock + row) * lda + k0 + ch,
                   &sA[buf][row * BK + ch]);
      }
    }
  };
  auto stageB = [&](int buf, int k0) {
    #pragma unroll
    for (int it = 0; it < (BN * 4) / 256; ++it) {
      int i = tid + it * 256;
      int row = i >> 2, ch = (i & 3) * 8;
      async_cp16(Bb + (size_t)(nBlock + row) * ldb + k0 + ch,
                 &sB[buf][row * BK + ch]);
    }
  };

  stageA(0, 0);
  stageB(0, 0);

  int buf = 0;
  for (int k0 = 0; k0 < K; k0 += BK, buf ^= 1) {
    wait_async0();        // own async copies into `buf` complete
    __syncthreads();      // every wave's staging of `buf` visible
    if (k0 + BK < K) {    // prefetch next slab while computing this one
      stageA(buf ^ 1, k0 + BK);
      stageB(buf ^ 1, k0 + BK);
    }

    const unsigned short* curA = sA[buf];
    const unsigned short* curB = sB[buf];
    union Frag { v16bf v; v8bf h[2]; };
    Frag afr[TM], bfr[TN];
    #pragma unroll
    for (int i = 0; i < TM; ++i) {
      int r = ((wm * TM + i) * 16 + lrow) * BK + kb;
      afr[i].h[0] = *(const v8bf*)&curA[r];
      afr[i].h[1] = *(const v8bf*)&curA[r + 16];
    }
    #pragma unroll
    for (int j = 0; j < TN; ++j) {
      int c = ((wn * TN + j) * 16 + lrow) * BK + kb;
      bfr[j].h[0] = *(const v8bf*)&curB[c];
      bfr[j].h[1] = *(const v8bf*)&curB[c + 16];
    }
    #pragma unroll
    for (int i = 0; i < TM; ++i)
      #pragma unroll
      for (int j = 0; j < TN; ++j)
        acc[i][j] = __builtin_amdgcn_wmma_f32_16x16x32_bf16(
            false, afr[i].v, false, bfr[j].v, (short)0, acc[i][j], false, false);
    __syncthreads();      // all reads of `buf` done before it is restaged
  }

  // ---- epilogue: C/D layout is lane=(n, m-half), reg r = m row ----
  #pragma unroll
  for (int i = 0; i < TM; ++i) {
    int mG0 = mBlock + (wm * TM + i) * 16 + lhalf * 8;
    #pragma unroll
    for (int j = 0; j < TN; ++j) {
      int nG = nBlock + (wn * TN + j) * 16 + lrow;
      float badd = 0.0f;
      if constexpr (OUT_MODE != 1) { if (bias) badd = bias[nG]; }
      #pragma unroll
      for (int r = 0; r < 8; ++r) {
        size_t off = (size_t)z * strideC + (size_t)(mG0 + r) * ldc + nG;
        float v = acc[i][j][r];
        if constexpr (OUT_MODE == 0) {
          ((unsigned short*)Cglob)[off] = f2bf(v + badd);
        } else if constexpr (OUT_MODE == 1) {
          ((float*)Cglob)[off] = v * scale;
        } else {
          ((float*)Cglob)[off] =
              v + badd + resid[(size_t)(mG0 + r) * ldc + nG];
        }
      }
    }
  }
}

// ---------------- stable row softmax over 2048 cols, in place ---------------
__global__ __launch_bounds__(256) void softmax_rows(float* __restrict__ att) {
  __shared__ float red[256];
  const int tid = threadIdx.x;
  float* p = att + (size_t)blockIdx.x * 2048 + tid * 8;
  float4 a = ((float4*)p)[0], b = ((float4*)p)[1];
  float x[8] = {a.x, a.y, a.z, a.w, b.x, b.y, b.z, b.w};
  float m = x[0];
  #pragma unroll
  for (int r = 1; r < 8; ++r) m = fmaxf(m, x[r]);
  red[tid] = m; __syncthreads();
  for (int s = 128; s > 0; s >>= 1) {
    if (tid < s) red[tid] = fmaxf(red[tid], red[tid + s]);
    __syncthreads();
  }
  m = red[0]; __syncthreads();
  float sum = 0.0f;
  #pragma unroll
  for (int r = 0; r < 8; ++r) { x[r] = __expf(x[r] - m); sum += x[r]; }
  red[tid] = sum; __syncthreads();
  for (int s = 128; s > 0; s >>= 1) {
    if (tid < s) red[tid] += red[tid + s];
    __syncthreads();
  }
  float inv = 1.0f / red[0];
  ((float4*)p)[0] = make_float4(x[0]*inv, x[1]*inv, x[2]*inv, x[3]*inv);
  ((float4*)p)[1] = make_float4(x[4]*inv, x[5]*inv, x[6]*inv, x[7]*inv);
}

// ---------------- LayerNorm over D=1024 -------------------------------------
__global__ __launch_bounds__(256) void layernorm_rows(
    const float* __restrict__ X, const float* __restrict__ gamma,
    const float* __restrict__ beta, float* __restrict__ out) {
  __shared__ float red[256];
  const int tid = threadIdx.x;
  const size_t row = blockIdx.x;
  float4 v = *(const float4*)(X + row * 1024 + tid * 4);
  red[tid] = v.x + v.y + v.z + v.w; __syncthreads();
  for (int s = 128; s > 0; s >>= 1) {
    if (tid < s) red[tid] += red[tid + s];
    __syncthreads();
  }
  float mean = red[0] * (1.0f / 1024.0f); __syncthreads();
  float d0 = v.x - mean, d1 = v.y - mean, d2 = v.z - mean, d3 = v.w - mean;
  red[tid] = d0*d0 + d1*d1 + d2*d2 + d3*d3; __syncthreads();
  for (int s = 128; s > 0; s >>= 1) {
    if (tid < s) red[tid] += red[tid + s];
    __syncthreads();
  }
  float rs = rsqrtf(red[0] * (1.0f / 1024.0f) + 1e-5f);
  int c = tid * 4;
  float4 g = *(const float4*)(gamma + c);
  float4 be = *(const float4*)(beta + c);
  float4 o = make_float4(d0*rs*g.x + be.x, d1*rs*g.y + be.y,
                         d2*rs*g.z + be.z, d3*rs*g.w + be.w);
  *(float4*)(out + row * 1024 + c) = o;
}

// ---------------------------------------------------------------------------
extern "C" void kernel_launch(void* const* d_in, const int* in_sizes, int n_in,
                              void* d_out, int out_size, void* d_ws, size_t ws_size,
                              hipStream_t stream) {
  (void)in_sizes; (void)n_in; (void)out_size; (void)ws_size;
  const float* key   = (const float*)d_in[0];
  const float* value = (const float*)d_in[1];
  const float* query = (const float*)d_in[2];
  const float* Wq = (const float*)d_in[3];
  const float* bq = (const float*)d_in[4];
  const float* Wk = (const float*)d_in[5];
  const float* bk = (const float*)d_in[6];
  const float* Wv = (const float*)d_in[7];
  const float* bv = (const float*)d_in[8];
  const float* Wo = (const float*)d_in[9];
  const float* bo = (const float*)d_in[10];
  const float* gamma = (const float*)d_in[11];
  const float* beta  = (const float*)d_in[12];

  constexpr int L = 2048, D = 1024, DPH = 64;
  constexpr int Mtok = 4096;            // B*L
  constexpr int NH   = 32;              // B*H contiguous head blocks
  const size_t nTok = (size_t)Mtok * D; // 4,194,304

  float* outO = (float*)d_out;
  float* att  = (float*)d_out + nTok;   // [32, 2048, 2048]

  // workspace carve (~88 MB, everything 16B-aligned)
  uint8_t* w = (uint8_t*)d_ws;
  auto aus = [&](size_t n) { unsigned short* p = (unsigned short*)w; w += n * 2; return p; };
  auto af  = [&](size_t n) { float* p = (float*)w; w += n * 4; return p; };
  unsigned short* q_bf  = aus(nTok);
  unsigned short* k_bf  = aus(nTok);
  unsigned short* v_bf  = aus(nTok);
  unsigned short* Wq_bf = aus((size_t)D * D);
  unsigned short* Wk_bf = aus((size_t)D * D);
  unsigned short* Wv_bf = aus((size_t)D * D);
  unsigned short* Wo_bf = aus((size_t)D * D);
  unsigned short* Q_bf  = aus(nTok);
  unsigned short* K_bf  = aus(nTok);
  unsigned short* V_bf  = aus(nTok);
  unsigned short* Vt_bf = aus(nTok);
  unsigned short* C_bf  = aus(nTok);
  float* Xres = af(nTok);

  dim3 blk(256);
  auto cvt = [&](const float* s, unsigned short* d, size_t n) {
    cvt_f32_to_bf16<<<dim3((unsigned)(n / 4 / 256)), blk, 0, stream>>>(s, d, (int)(n / 4));
  };
  cvt(query, q_bf, nTok); cvt(key, k_bf, nTok); cvt(value, v_bf, nTok);
  cvt(Wq, Wq_bf, (size_t)D * D); cvt(Wk, Wk_bf, (size_t)D * D);
  cvt(Wv, Wv_bf, (size_t)D * D); cvt(Wo, Wo_bf, (size_t)D * D);

  // Q/K/V projections: [4096x1024] = X * W^T + b  -> bf16
  gemm_wmma<false, 128, 0><<<dim3(D / 128, Mtok / 128, 1), blk, 0, stream>>>(
      q_bf, Wq_bf, bq, nullptr, Q_bf, D, D, D, D, 0, 0, 0, 1.0f);
  gemm_wmma<false, 128, 0><<<dim3(D / 128, Mtok / 128, 1), blk, 0, stream>>>(
      k_bf, Wk_bf, bk, nullptr, K_bf, D, D, D, D, 0, 0, 0, 1.0f);
  gemm_wmma<false, 128, 0><<<dim3(D / 128, Mtok / 128, 1), blk, 0, stream>>>(
      v_bf, Wv_bf, bv, nullptr, V_bf, D, D, D, D, 0, 0, 0, 1.0f);

  transpose_heads_v<<<dim3((unsigned)(nTok / 256)), blk, 0, stream>>>(V_bf, Vt_bf);

  // scores: per head z, S = Qz * Kz^T * 0.5 -> f32 attention buffer
  gemm_wmma<false, 128, 1><<<dim3(L / 128, L / 128, NH), blk, 0, stream>>>(
      Q_bf, K_bf, nullptr, nullptr, att, DPH, DPH, DPH, L,
      (long long)L * DPH, (long long)L * DPH, (long long)L * L, 0.5f);

  softmax_rows<<<dim3(NH * L), blk, 0, stream>>>(att);

  // context: Cz = Az * Vz  (A f32 -> bf16 in LDS stage; B = Vt [64 x 2048])
  gemm_wmma<true, 64, 0><<<dim3(1, L / 128, NH), blk, 0, stream>>>(
      att, Vt_bf, nullptr, nullptr, C_bf, L, L, L, DPH,
      (long long)L * L, (long long)DPH * L, (long long)L * DPH, 1.0f);

  // output projection + bias + residual -> f32
  gemm_wmma<false, 128, 2><<<dim3(D / 128, Mtok / 128, 1), blk, 0, stream>>>(
      C_bf, Wo_bf, bo, query, Xres, D, D, D, D, 0, 0, 0, 1.0f);

  layernorm_rows<<<dim3(Mtok), blk, 0, stream>>>(Xres, gamma, beta, outO);
}
